// SelectedUnitsHead_36713380446756
// MI455X (gfx1250) — matrix operand order repaired and checked by reading, not compiled, not run on hardware
//
#include <hip/hip_runtime.h>
#include <hip/hip_bf16.h>

// ---------------- problem constants ----------------
#define BSZ        256
#define NENT       512
#define NP1        513       // N + 1 (end flag row)
#define E_DIM      256
#define KEY_DIM    32
#define FUNC_DIM   256
#define INPUT_DIM  1024
#define HIDDEN     32
#define GATES      128       // 4*HIDDEN
#define UT_DIM     259
#define MAX_STEPS  64
#define NEGV       1.0e9f
#define EPSV       1e-5f
#define KE_PITCH   33        // LDS pitch for key rows (bank-conflict free)

typedef __attribute__((ext_vector_type(16))) _Float16 v16h;
typedef __attribute__((ext_vector_type(8)))  float    v8f;
typedef unsigned int u32x4 __attribute__((ext_vector_type(4)));
typedef int          i32x8 __attribute__((ext_vector_type(8)));
typedef int          i32x4 __attribute__((ext_vector_type(4)));

// Tensor Data Mover availability (builtin arity differs across toolchains)
#if defined(__has_builtin)
#if __has_builtin(__builtin_amdgcn_tensor_load_to_lds) && __has_builtin(__builtin_amdgcn_s_wait_tensorcnt)
#define USE_TDM 1
#endif
#endif
#ifndef USE_TDM
#define USE_TDM 0
#endif

__device__ __forceinline__ float sigmoidf_(float x) { return 1.0f / (1.0f + __expf(-x)); }

// ======================================================================
// Kernel 1: key_e = entity_embedding @ W_key + b_key   via v_wmma_f32_16x16x32_f16
//   A: [131072, 256] f32 -> f16 fragments, W: [256, 32], out: key_e [B, 513, 32]
//   One wave computes a 16-row x 32-col tile (two 16x16 WMMA accumulators).
// ======================================================================
__global__ __launch_bounds__(256) void key_gemm_wmma(
    const float* __restrict__ A,    // [B*N, E_DIM]
    const float* __restrict__ Wk,   // [E_DIM, KEY_DIM]
    const float* __restrict__ bk,   // [KEY_DIM]
    float* __restrict__ key_e)      // [B, NP1, KEY_DIM]
{
    const int lane   = threadIdx.x & 31;
    const int wave   = threadIdx.x >> 5;
    const int tile_m = blockIdx.x * 8 + wave;          // 16-row tile index
    const int m_loc  = lane & 15;
    const int khalf  = (lane >> 4) * 8;                // A-frag K half select
    const int kbB    = (lane >> 4) * 16;               // B-frag K half select
    const int nl     = lane & 15;
    const size_t rowbase = (size_t)(tile_m * 16 + m_loc) * E_DIM;

    v8f c0 = {};
    v8f c1 = {};

    for (int k0 = 0; k0 < E_DIM; k0 += 32) {
        // ---- A fragment: 16-bit 16x32 layout (ISA 7.12.2); 64-bit pair loads ----
        const float2* Arow = (const float2*)(A + rowbase + k0);
        v16h a;
#pragma unroll
        for (int j = 0; j < 8; ++j) {
            int k = (j < 4 ? 0 : 16) + khalf + 2 * (j & 3);
            float2 p = Arow[k >> 1];
            a[2 * j]     = (_Float16)p.x;
            a[2 * j + 1] = (_Float16)p.y;
        }
        // ---- B fragments (two 16-col tiles): row K striped across lanes ----
        v16h b0, b1;
#pragma unroll
        for (int j = 0; j < 8; ++j) {
            int k = k0 + (j < 4 ? 0 : 16) + khalf + 2 * (j & 3);
            b0[2 * j]     = (_Float16)Wk[(size_t)k * KEY_DIM + nl];
            b0[2 * j + 1] = (_Float16)Wk[(size_t)(k + 1) * KEY_DIM + nl];
            b1[2 * j]     = (_Float16)Wk[(size_t)k * KEY_DIM + 16 + nl];
            b1[2 * j + 1] = (_Float16)Wk[(size_t)(k + 1) * KEY_DIM + 16 + nl];
        }
        c0 = __builtin_amdgcn_wmma_f32_16x16x32_f16(false, a, false, b0, (short)0, c0, false, false);
        c1 = __builtin_amdgcn_wmma_f32_16x16x32_f16(false, a, false, b1, (short)0, c1, false, false);
    }

    // ---- store: C layout VGPR r -> M = r (lanes 0-15) / r+8 (lanes 16-31) ----
#pragma unroll
    for (int r = 0; r < 8; ++r) {
        int mrow = (lane < 16) ? r : (r + 8);
        int gm   = tile_m * 16 + mrow;          // global row in [0, B*N)
        int b    = gm >> 9;                     // / NENT
        int row  = gm & (NENT - 1);
        float* o = key_e + ((size_t)b * NP1 + row) * KEY_DIM;
        o[nl]      = c0[r] + bk[nl];
        o[16 + nl] = c1[r] + bk[16 + nl];
    }
}

// zero the appended end-flag key row (row N of each batch)
__global__ void key_end_row_init(float* __restrict__ key_e)
{
    key_e[((size_t)blockIdx.x * NP1 + NENT) * KEY_DIM + threadIdx.x] = 0.0f;
}

// ======================================================================
// Kernel 2: initial query
//   s = relu(emb@W_fc1 + b_fc1) + relu(utm@W_func + b_func)  [B, 256]
//   x0 = relu(s @ W_fc2 + b_fc2)                              [B, 32]
// ======================================================================
__global__ __launch_bounds__(256) void init_query_kernel(
    const float* __restrict__ emb,   // [B, INPUT_DIM]
    const float* __restrict__ utm,   // [B, UT_DIM]
    const float* __restrict__ Wfu,   // [UT_DIM, FUNC_DIM]
    const float* __restrict__ bfu,   // [FUNC_DIM]
    const float* __restrict__ W1,    // [INPUT_DIM, FUNC_DIM]
    const float* __restrict__ b1,    // [FUNC_DIM]
    const float* __restrict__ W2,    // [FUNC_DIM, KEY_DIM]
    const float* __restrict__ b2,    // [KEY_DIM]
    float* __restrict__ x0)          // [B, KEY_DIM]
{
    __shared__ float s[FUNC_DIM];
    const int b = blockIdx.x;
    const int j = threadIdx.x;

    float a1 = b1[j];
    const float* e = emb + (size_t)b * INPUT_DIM;
    for (int k = 0; k < INPUT_DIM; ++k) a1 += e[k] * W1[(size_t)k * FUNC_DIM + j];

    float af = bfu[j];
    const float* u = utm + (size_t)b * UT_DIM;
    for (int k = 0; k < UT_DIM; ++k) af += u[k] * Wfu[(size_t)k * FUNC_DIM + j];

    s[j] = fmaxf(a1, 0.0f) + fmaxf(af, 0.0f);
    __syncthreads();

    if (j < KEY_DIM) {
        float a2 = b2[j];
        for (int k = 0; k < FUNC_DIM; ++k) a2 += s[k] * W2[(size_t)k * KEY_DIM + j];
        x0[(size_t)b * KEY_DIM + j] = fmaxf(a2, 0.0f);
    }
}

// ======================================================================
// Kernel 3: 64-step LN-LSTM pointer loop. One block per batch element.
//   key_e[b] staged into LDS (pitch 33) by the Tensor Data Mover: one
//   tensor_load_to_lds with a 2D D# (tile 32x513, 4B elems) whose pad
//   fields (interval=32 DWORDs, amount=1 DWORD) produce the pitch-33
//   bank-conflict-free layout in hardware. Tracked by TENSORcnt.
// ======================================================================
__global__ __launch_bounds__(256) void lstm_loop_kernel(
    const float* __restrict__ key_e,   // [B, NP1, KEY_DIM]
    const float* __restrict__ x0,      // [B, KEY_DIM]
    const float* __restrict__ mask_in, // [B, NENT]
    const float* __restrict__ wx,      // [KEY_DIM, GATES]
    const float* __restrict__ wh,      // [HIDDEN, GATES]
    const float* __restrict__ b_lstm,  // [GATES]
    const float* __restrict__ ln_gx, const float* __restrict__ ln_bx,
    const float* __restrict__ ln_gh, const float* __restrict__ ln_bh,
    float* __restrict__ out_logits,    // [MAX_STEPS, B, NP1]
    float* __restrict__ out_valid,     // [MAX_STEPS, B]
    float* __restrict__ out_units)     // [B, NP1]
{
    extern __shared__ float smem[];
    float* ke   = smem;                    // NP1 * KE_PITCH
    float* mm   = ke + NP1 * KE_PITCH;     // NP1
    float* uni  = mm + NP1;                // NP1
    float* hb   = uni + NP1;               // HIDDEN
    float* cb   = hb + HIDDEN;             // HIDDEN
    float* xin  = cb + HIDDEN;             // KEY_DIM
    float* gx   = xin + KEY_DIM;           // GATES (later holds final gates)
    float* gh   = gx + GATES;              // GATES
    float* red  = gh + GATES;              // 512
    int*   redi = (int*)(red + 512);       // 256
    int*   acti = redi + 256;              // 1

    const int b   = blockIdx.x;
    const int tid = threadIdx.x;
    const float* keg = key_e + (size_t)b * NP1 * KEY_DIM;

#if USE_TDM
    // --- TDM: DMA key_e[b] (513x32 f32) into LDS with pitch-33 padding ---
    if (tid == 0) {
        unsigned long long ga = (unsigned long long)(size_t)keg;
        unsigned lds_addr = (unsigned)(size_t)ke;   // low 32 bits = LDS offset
        u32x4 g0;
        g0.x = 1u;                                        // count=1 valid descriptor
        g0.y = lds_addr;                                  // D# lds_addr
        g0.z = (unsigned)(ga & 0xFFFFFFFFu);              // global_addr[31:0]
        g0.w = (unsigned)((ga >> 32) & 0x01FFFFFFu)       // global_addr[56:32]
             | 0x80000000u;                               // type=2 ("image")
        i32x8 g1;
        g1[0] = (2 << 16)        // data_size = 4 bytes
              | (1 << 20)        // pad_enable
              | (4 << 22)        // pad_interval: 32 DWORDs (128B tile row)
              | (0 << 25);       // pad_amount: 1 DWORD -> pitch 33
        g1[1] = (int)((unsigned)KEY_DIM << 16);           // tensor_dim0[15:0]
        g1[2] = (int)((unsigned)NP1 << 16);               // dim0 hi=0 | tensor_dim1[15:0]
        g1[3] = (int)((unsigned)KEY_DIM << 16);           // dim1 hi=0 | tile_dim0=32
        g1[4] = NP1;                                      // tile_dim1=513, tile_dim2=0
        g1[5] = KEY_DIM;                                  // tensor_dim0_stride[31:0]
        g1[6] = 0;                                        // stride0 hi | dim1_stride lo
        g1[7] = 0;
        i32x4 gz = {0, 0, 0, 0};
#if defined(__clang_major__) && __clang_major__ >= 23
        i32x8 z8 = {0, 0, 0, 0, 0, 0, 0, 0};
        __builtin_amdgcn_tensor_load_to_lds(g0, g1, gz, gz, z8, 0);
#else
        __builtin_amdgcn_tensor_load_to_lds(g0, g1, gz, gz, 0);
#endif
    }
    __builtin_amdgcn_s_wait_tensorcnt(0);
#else
    for (int idx = tid; idx < NP1 * KEY_DIM; idx += 256) {
        int n = idx >> 5, k = idx & 31;
        ke[n * KE_PITCH + k] = keg[idx];
    }
#endif

    for (int n = tid; n < NP1; n += 256) {
        mm[n]  = (n < NENT) ? mask_in[(size_t)b * NENT + n] : 1.0f;
        uni[n] = 0.0f;
    }
    if (tid < HIDDEN) { hb[tid] = 0.0f; cb[tid] = 0.0f; }
    if (tid < KEY_DIM) xin[tid] = x0[(size_t)b * KEY_DIM + tid];
    if (tid == 0) acti[0] = 1;
    __syncthreads();

    for (int t = 0; t < MAX_STEPS; ++t) {
        // ---- gate matvecs: gx = xin@wx, gh = h@wh ----
        if (tid < GATES) {
            float ax = 0.0f, ah = 0.0f;
            for (int k = 0; k < KEY_DIM; ++k) {
                ax += xin[k] * wx[k * GATES + tid];
                ah += hb[k]  * wh[k * GATES + tid];
            }
            gx[tid] = ax; gh[tid] = ah;
        }
        __syncthreads();

        // ---- dual LayerNorm: means ----
        red[tid]       = (tid < GATES) ? gx[tid] : 0.0f;
        red[256 + tid] = (tid < GATES) ? gh[tid] : 0.0f;
        __syncthreads();
        for (int s = 128; s > 0; s >>= 1) {
            if (tid < s) { red[tid] += red[tid + s]; red[256 + tid] += red[256 + tid + s]; }
            __syncthreads();
        }
        float mux = red[0]   * (1.0f / GATES);
        float muh = red[256] * (1.0f / GATES);
        __syncthreads();
        // ---- variances ----
        float dx = (tid < GATES) ? (gx[tid] - mux) : 0.0f;
        float dh = (tid < GATES) ? (gh[tid] - muh) : 0.0f;
        red[tid] = dx * dx; red[256 + tid] = dh * dh;
        __syncthreads();
        for (int s = 128; s > 0; s >>= 1) {
            if (tid < s) { red[tid] += red[tid + s]; red[256 + tid] += red[256 + tid + s]; }
            __syncthreads();
        }
        float rvx = rsqrtf(red[0]   * (1.0f / GATES) + EPSV);
        float rvh = rsqrtf(red[256] * (1.0f / GATES) + EPSV);
        __syncthreads();
        if (tid < GATES) {
            float nx = dx * rvx * ln_gx[tid] + ln_bx[tid];
            float nh = dh * rvh * ln_gh[tid] + ln_bh[tid];
            gx[tid] = nx + nh + b_lstm[tid];       // final gate pre-activations
        }
        __syncthreads();

        // ---- LSTM cell (i,f,o,u) ----
        if (tid < HIDDEN) {
            float ig = gx[tid], fg = gx[HIDDEN + tid];
            float og = gx[2 * HIDDEN + tid], ug = gx[3 * HIDDEN + tid];
            float c2 = sigmoidf_(fg) * cb[tid] + sigmoidf_(ig) * tanhf(ug);
            float h2 = sigmoidf_(og) * tanhf(c2);
            cb[tid] = c2; hb[tid] = h2; xin[tid] = h2;
        }
        __syncthreads();

        // ---- qr over 513 entities + first-max argmax ----
        float bestv = -INFINITY; int besti = 1 << 30;
        float* lrow = out_logits + ((size_t)t * BSZ + b) * NP1;
        for (int n = tid; n < NP1; n += 256) {
            const float* kr = ke + n * KE_PITCH;
            float acc = 0.0f;
#pragma unroll
            for (int k = 0; k < KEY_DIM; ++k) acc += hb[k] * kr[k];
            float q = acc * (1.0f / KEY_DIM) - (1.0f - mm[n]) * NEGV;
            lrow[n] = q;
            if (q > bestv) { bestv = q; besti = n; }   // ascending n -> first max kept
        }
        red[tid] = bestv; redi[tid] = besti;
        __syncthreads();
        for (int s = 128; s > 0; s >>= 1) {
            if (tid < s) {
                float v2 = red[tid + s]; int i2 = redi[tid + s];
                if (v2 > red[tid] || (v2 == red[tid] && i2 < redi[tid])) {
                    red[tid] = v2; redi[tid] = i2;
                }
            }
            __syncthreads();
        }

        // ---- selection bookkeeping ----
        if (tid == 0) {
            int ent = redi[0];
            out_valid[(size_t)t * BSZ + b] = acti[0] ? 1.0f : 0.0f;   // incoming 'active'
            int sel_end = (ent == NENT);
            if (acti[0] && !sel_end) { uni[ent] = 1.0f; mm[ent] = 0.0f; }
            if (sel_end) acti[0] = 0;
        }
        __syncthreads();
    }

    for (int n = tid; n < NP1; n += 256)
        out_units[(size_t)b * NP1 + n] = uni[n];
}

// ======================================================================
// Kernel 4: sel = mean_n(units * key_e); new_emb = emb + sel@W_embed + b_embed
// ======================================================================
__global__ __launch_bounds__(256) void finalize_kernel(
    const float* __restrict__ units,   // [B, NP1] (from d_out)
    const float* __restrict__ key_e,   // [B, NP1, KEY_DIM]
    const float* __restrict__ emb,     // [B, INPUT_DIM]
    const float* __restrict__ We,      // [KEY_DIM, INPUT_DIM]
    const float* __restrict__ be,      // [INPUT_DIM]
    float* __restrict__ out_emb)       // [B, INPUT_DIM]
{
    __shared__ float sel[KEY_DIM];
    const int b = blockIdx.x;
    const int tid = threadIdx.x;

    if (tid < KEY_DIM) {
        float acc = 0.0f;
        const float* ub = units + (size_t)b * NP1;
        const float* kb = key_e + (size_t)b * NP1 * KEY_DIM;
        for (int n = 0; n < NP1; ++n) acc += ub[n] * kb[n * KEY_DIM + tid];
        sel[tid] = acc * (1.0f / NP1);
    }
    __syncthreads();

    for (int j = tid; j < INPUT_DIM; j += 256) {
        float acc = be[j];
#pragma unroll
        for (int k = 0; k < KEY_DIM; ++k) acc += sel[k] * We[k * INPUT_DIM + j];
        out_emb[(size_t)b * INPUT_DIM + j] = emb[(size_t)b * INPUT_DIM + j] + acc;
    }
}

// ======================================================================
extern "C" void kernel_launch(void* const* d_in, const int* in_sizes, int n_in,
                              void* d_out, int out_size, void* d_ws, size_t ws_size,
                              hipStream_t stream) {
    const float* emb    = (const float*)d_in[0];
    const float* utm    = (const float*)d_in[1];
    const float* amask  = (const float*)d_in[2];
    const float* ent    = (const float*)d_in[3];
    const float* W_key  = (const float*)d_in[4];
    const float* b_key  = (const float*)d_in[5];
    const float* W_func = (const float*)d_in[6];
    const float* b_func = (const float*)d_in[7];
    const float* W_fc1  = (const float*)d_in[8];
    const float* b_fc1  = (const float*)d_in[9];
    const float* W_fc2  = (const float*)d_in[10];
    const float* b_fc2  = (const float*)d_in[11];
    const float* W_emb  = (const float*)d_in[12];
    const float* b_emb  = (const float*)d_in[13];
    const float* wx     = (const float*)d_in[14];
    const float* wh     = (const float*)d_in[15];
    const float* b_lstm = (const float*)d_in[16];
    const float* ln_gx  = (const float*)d_in[17];
    const float* ln_bx  = (const float*)d_in[18];
    const float* ln_gh  = (const float*)d_in[19];
    const float* ln_bh  = (const float*)d_in[20];

    // workspace layout
    float* key_e = (float*)d_ws;                                  // B*NP1*KEY_DIM
    float* x0    = key_e + (size_t)BSZ * NP1 * KEY_DIM;           // B*KEY_DIM

    // output layout (tuple flattened in return order, all f32)
    float* out_logits = (float*)d_out;                                      // 64*B*513
    float* out_valid  = out_logits + (size_t)MAX_STEPS * BSZ * NP1;         // 64*B
    float* out_units  = out_valid + (size_t)MAX_STEPS * BSZ;                // B*513
    float* out_emb    = out_units + (size_t)BSZ * NP1;                      // B*1024

    // 1) key_e GEMM on the WMMA pipe  (M tiles = 131072/16 = 8192; 8 waves/block)
    key_gemm_wmma<<<1024, 256, 0, stream>>>(ent, W_key, b_key, key_e);
    key_end_row_init<<<BSZ, KEY_DIM, 0, stream>>>(key_e);

    // 2) initial query
    init_query_kernel<<<BSZ, 256, 0, stream>>>(emb, utm, W_func, b_func,
                                               W_fc1, b_fc1, W_fc2, b_fc2, x0);

    // 3) sequential pointer loop (one workgroup per batch element)
    size_t shmem = (size_t)(NP1 * KE_PITCH + NP1 + NP1 + HIDDEN + HIDDEN + KEY_DIM
                            + GATES + GATES + 512) * sizeof(float)
                 + (size_t)(256 + 1) * sizeof(int);
    lstm_loop_kernel<<<BSZ, 256, shmem, stream>>>(key_e, x0, amask, wx, wh, b_lstm,
                                                  ln_gx, ln_bx, ln_gh, ln_bh,
                                                  out_logits, out_valid, out_units);

    // 4) selected embedding
    finalize_kernel<<<BSZ, 256, 0, stream>>>(out_units, key_e, emb, W_emb, b_emb, out_emb);
}
